// EGGNEncoder_67688684585222
// MI455X (gfx1250) — compile-verified, matrix operand order checked by reading, not run on hardware
//
#include <hip/hip_runtime.h>
#include <hip/hip_bf16.h>

typedef __bf16 bf16_t;
typedef __attribute__((ext_vector_type(16))) bf16_t v16bf;
typedef __attribute__((ext_vector_type(8)))  float  v8f;

#define HD      64
#define NLAYERS 2
#define WAVES   8
#define MAXBLK  1536

#define WB() __builtin_amdgcn_wave_barrier()

// ---------- helpers ----------

static __device__ __forceinline__ unsigned short f2bf(float f) {
    union { float f; unsigned int u; } c; c.f = f;
    unsigned int u = c.u;
    return (unsigned short)((u + 0x7FFFu + ((u >> 16) & 1u)) >> 16); // RNE
}

static __device__ __forceinline__ float silu_f(float v) {
    return v / (1.0f + __expf(-v));
}

union FragU {
    v16bf v;
    uint4 q[2];
    unsigned short s[16];
};

static __device__ __forceinline__ v8f wmma_bf16(v16bf a, v16bf b, v8f c) {
    // (neg_a, A, neg_b, B, c_mod, C, reuse_a, reuse_b)
    return __builtin_amdgcn_wmma_f32_16x16x32_bf16(false, a, false, b, (short)0, c,
                                                   false, false);
}

// B-fragment (K=32 x N=16 tile): 32 lanes x 16 bf16 contiguous, lane-major.
// fr may point to LDS (staged weights) or global.
static __device__ __forceinline__ FragU load_bfrag(const unsigned short* fr, int fid, int lane) {
    FragU b;
    const uint4* p = reinterpret_cast<const uint4*>(fr + fid * 512) + lane * 2;
    b.q[0] = p[0];
    b.q[1] = p[1];
    return b;
}

// ---------- weight packing ----------
// Per layer, 56 fragments of 512 bf16:
//   fid  0..15 : e_w1 rows 0..127 (kc=fid>>2, nb=fid&3)
//   fid 16..23 : e_w2             (kc 0..1)
//   fid 24..31 : x_w1
//   fid 32..47 : h_w1 rows 0..127
//   fid 48..55 : h_w2
__global__ void prep_weights_k(const float* __restrict__ e_w1, const float* __restrict__ e_w2,
                               const float* __restrict__ x_w1, const float* __restrict__ h_w1,
                               const float* __restrict__ h_w2, unsigned short* __restrict__ frags)
{
    const int bid = blockIdx.x;          // NLAYERS*56 blocks
    const int l   = bid / 56;
    const int fid = bid % 56;
    const float* W; int kc, nb;
    if (fid < 16)      { W = e_w1 + (size_t)l * 129 * HD;    kc = fid >> 2;        nb = fid & 3; }
    else if (fid < 24) { int f = fid - 16; W = e_w2 + (size_t)l * HD * HD;     kc = f >> 2; nb = f & 3; }
    else if (fid < 32) { int f = fid - 24; W = x_w1 + (size_t)l * HD * HD;     kc = f >> 2; nb = f & 3; }
    else if (fid < 48) { int f = fid - 32; W = h_w1 + (size_t)l * 2 * HD * HD; kc = f >> 2; nb = f & 3; }
    else               { int f = fid - 48; W = h_w2 + (size_t)l * HD * HD;     kc = f >> 2; nb = f & 3; }
    unsigned short* dst = frags + (size_t)bid * 512;
    for (int t = threadIdx.x; t < 512; t += blockDim.x) {
        const int lane = t >> 4, i = t & 15;
        const int v = i >> 1, j = i & 1;
        const int kl = ((v >= 4) ? 16 : 0) + 8 * (lane >> 4) + 2 * (v & 3) + j;
        const int k  = kc * 32 + kl;
        const int n  = nb * 16 + (lane & 15);
        dst[t] = f2bf(W[(size_t)k * HD + n]);
    }
}

__global__ void prep_small_k(const float* __restrict__ e_w1, const float* __restrict__ x_w2,
                             float* __restrict__ ew1_last, float* __restrict__ xw2buf)
{
    const int t = threadIdx.x;           // 0..127
    if (t < NLAYERS * HD) {
        const int l = t >> 6, n = t & 63;
        ew1_last[t] = e_w1[(size_t)l * 129 * HD + 128 * HD + n]; // row 128 (r_ij column)
        xw2buf[t]   = x_w2[(size_t)l * HD + n];
    }
}

// ---------- embedding + per-layer init ----------

__global__ void embed_k(const int* __restrict__ an, const float* __restrict__ emb,
                        const float* __restrict__ pos, float* __restrict__ h,
                        unsigned short* __restrict__ hbf, float* __restrict__ xA, int N)
{
    const int i = blockIdx.x * blockDim.x + threadIdx.x;
    if (i < N * HD) {
        const int n = i / HD;
        const float v = emb[(size_t)an[n] * HD + (i % HD)];
        h[i]   = v;
        hbf[i] = f2bf(v);
    }
    if (i < N * 3) xA[i] = pos[i];
}

__global__ void init_layer_k(float* __restrict__ mi, const float* __restrict__ xcur,
                             float* __restrict__ xnext, int N)
{
    const int i = blockIdx.x * blockDim.x + threadIdx.x;
    if (i < N * HD) mi[i] = 0.0f;
    if (i < N * 3)  xnext[i] = xcur[i];
}

// ---------- edge kernel: all three edge MLPs + scatter ----------
// Weights (fids 0..31) staged in LDS once per block; each wave then grid-strides
// over 16-edge tiles with a uniform iteration count (barrier-safe).

__global__ __launch_bounds__(32 * WAVES)
void edge_k(const int* __restrict__ ei, const float* __restrict__ xcur,
            const unsigned short* __restrict__ hbf, const unsigned short* __restrict__ fr,
            const float* __restrict__ eb1, const float* __restrict__ eb2,
            const float* __restrict__ xb1, const float* __restrict__ ew1_last,
            const float* __restrict__ xw2, const float* __restrict__ xb2p,
            float* __restrict__ xnext, float* __restrict__ mi, int E, int iters)
{
    __shared__ __align__(16) unsigned short wfrag[32 * 512];          // 32 KB weights
    __shared__ __align__(16) unsigned short tile_s[WAVES][16 * HD];   // per-wave staging
    __shared__ float rbuf[WAVES][16];
    __shared__ float wbuf[WAVES][16];
    __shared__ int   dstb[WAVES][16];

    const int wave = threadIdx.x >> 5;
    const int lane = threadIdx.x & 31;
    const int half = lane >> 4;
    const int m    = lane & 15;

    // cooperative weight stage: global -> LDS (2048 uint4)
    {
        const uint4* s4 = reinterpret_cast<const uint4*>(fr);
        uint4* d4 = reinterpret_cast<uint4*>(wfrag);
        #pragma unroll
        for (int i = 0; i < (32 * 512 / 8) / (32 * WAVES); ++i)
            d4[threadIdx.x + i * 32 * WAVES] = s4[threadIdx.x + i * 32 * WAVES];
    }

    // hoist loop-invariant per-lane scalars
    float c_eb1[4], c_eb2[4], c_xb1[4], c_wl[4], c_xw[4];
    #pragma unroll
    for (int nb = 0; nb < 4; ++nb) {
        c_eb1[nb] = eb1[nb * 16 + m];
        c_eb2[nb] = eb2[nb * 16 + m];
        c_xb1[nb] = xb1[nb * 16 + m];
        c_wl[nb]  = ew1_last[nb * 16 + m];
        c_xw[nb]  = xw2[nb * 16 + m];
    }
    const float xb2 = xb2p[0];

    __syncthreads();   // weights visible to all waves

    const int stride = gridDim.x * WAVES;
    const int wtile0 = blockIdx.x * WAVES + wave;

    for (int it = 0; it < iters; ++it) {
        const int tile = wtile0 + it * stride;
        const int e    = tile * 16 + m;
        const bool ev  = (e < E);
        const int ec   = ev ? e : 0;

        const int src = ei[ec];
        const int dst = ei[E + ec];
        const float dx = xcur[src * 3 + 0] - xcur[dst * 3 + 0];
        const float dy = xcur[src * 3 + 1] - xcur[dst * 3 + 1];
        const float dz = xcur[src * 3 + 2] - xcur[dst * 3 + 2];
        const float r  = sqrtf(dx * dx + dy * dy + dz * dz);
        if (half == 0) { rbuf[wave][m] = r; dstb[wave][m] = dst; }
        WB();

        // A fragments for h[dst] / h[src]: two K=32 chunks each (A layout, 16-bit)
        FragU Ad[2], As[2];
        {
            const uint4* hd = reinterpret_cast<const uint4*>(hbf + (size_t)dst * HD);
            const uint4* hs = reinterpret_cast<const uint4*>(hbf + (size_t)src * HD);
            #pragma unroll
            for (int kc = 0; kc < 2; ++kc) {
                Ad[kc].q[0] = hd[kc * 4 + half];
                Ad[kc].q[1] = hd[kc * 4 + half + 2];
                As[kc].q[0] = hs[kc * 4 + half];
                As[kc].q[1] = hs[kc * 4 + half + 2];
            }
        }

        // Stage 1: t1 = silu(feat @ e_w1 + e_b1), feat = [h_dst | h_src | r]
        #pragma unroll
        for (int nb = 0; nb < 4; ++nb) {
            v8f acc;
            #pragma unroll
            for (int v = 0; v < 8; ++v) acc[v] = c_eb1[nb];
            acc = wmma_bf16(Ad[0].v, load_bfrag(wfrag,  0 + nb, lane).v, acc);
            acc = wmma_bf16(Ad[1].v, load_bfrag(wfrag,  4 + nb, lane).v, acc);
            acc = wmma_bf16(As[0].v, load_bfrag(wfrag,  8 + nb, lane).v, acc);
            acc = wmma_bf16(As[1].v, load_bfrag(wfrag, 12 + nb, lane).v, acc);
            #pragma unroll
            for (int v = 0; v < 8; ++v) {      // rank-1 term for K=128 (r_ij)
                const float t = acc[v] + rbuf[wave][v + 8 * half] * c_wl[nb];
                tile_s[wave][(v + 8 * half) * HD + nb * 16 + m] = f2bf(silu_f(t));
            }
        }
        WB();

        // Stage 2: m_ij = silu(t1 @ e_w2 + e_b2)
        FragU At[2];
        {
            const uint4* ts = reinterpret_cast<const uint4*>(&tile_s[wave][m * HD]);
            #pragma unroll
            for (int kc = 0; kc < 2; ++kc) {
                At[kc].q[0] = ts[kc * 4 + half];
                At[kc].q[1] = ts[kc * 4 + half + 2];
            }
        }
        v8f mfrag[4];
        #pragma unroll
        for (int nb = 0; nb < 4; ++nb) {
            v8f acc;
            #pragma unroll
            for (int v = 0; v < 8; ++v) acc[v] = c_eb2[nb];
            acc = wmma_bf16(At[0].v, load_bfrag(wfrag, 16 + nb, lane).v, acc);
            acc = wmma_bf16(At[1].v, load_bfrag(wfrag, 20 + nb, lane).v, acc);
            #pragma unroll
            for (int v = 0; v < 8; ++v) mfrag[nb][v] = silu_f(acc[v]);
        }
        WB();
        #pragma unroll
        for (int nb = 0; nb < 4; ++nb)
            #pragma unroll
            for (int v = 0; v < 8; ++v)
                tile_s[wave][(v + 8 * half) * HD + nb * 16 + m] = f2bf(mfrag[nb][v]);
        WB();

        // Stage 3: p = silu(m @ x_w1 + x_b1); W = p @ x_w2 + x_b2 (cross-lane reduce)
        FragU Am[2];
        {
            const uint4* ts = reinterpret_cast<const uint4*>(&tile_s[wave][m * HD]);
            #pragma unroll
            for (int kc = 0; kc < 2; ++kc) {
                Am[kc].q[0] = ts[kc * 4 + half];
                Am[kc].q[1] = ts[kc * 4 + half + 2];
            }
        }
        float wpart[8];
        #pragma unroll
        for (int v = 0; v < 8; ++v) wpart[v] = 0.0f;
        #pragma unroll
        for (int nb = 0; nb < 4; ++nb) {
            v8f acc;
            #pragma unroll
            for (int v = 0; v < 8; ++v) acc[v] = c_xb1[nb];
            acc = wmma_bf16(Am[0].v, load_bfrag(wfrag, 24 + nb, lane).v, acc);
            acc = wmma_bf16(Am[1].v, load_bfrag(wfrag, 28 + nb, lane).v, acc);
            #pragma unroll
            for (int v = 0; v < 8; ++v) wpart[v] += silu_f(acc[v]) * c_xw[nb];
        }
        #pragma unroll
        for (int off = 1; off < 16; off <<= 1)
            #pragma unroll
            for (int v = 0; v < 8; ++v)
                wpart[v] += __shfl_xor(wpart[v], off, 32);
        if (m == 0) {
            #pragma unroll
            for (int v = 0; v < 8; ++v) wbuf[wave][v + 8 * half] = wpart[v] + xb2;
        }
        WB();

        // Scatter: coordinate update (lanes 0..15 each own one edge)
        if (half == 0 && ev) {
            const float W = wbuf[wave][m];
            atomicAdd(&xnext[(size_t)dst * 3 + 0], W * dx);
            atomicAdd(&xnext[(size_t)dst * 3 + 1], W * dy);
            atomicAdd(&xnext[(size_t)dst * 3 + 2], W * dz);
        }
        // Scatter: m_i segment sum (f32 atomics into L2-resident table)
        #pragma unroll
        for (int nb = 0; nb < 4; ++nb) {
            #pragma unroll
            for (int v = 0; v < 8; ++v) {
                const int M = v + 8 * half;
                if (tile * 16 + M < E)
                    atomicAdd(&mi[(size_t)dstb[wave][M] * HD + nb * 16 + m], mfrag[nb][v]);
            }
        }
        WB();   // keep rbuf/dstb/tile_s stable until all reads of this iter are done
    }
}

// ---------- node kernel: h += silu([h|m_i] @ h_w1 + b1) @ h_w2 + b2 ----------

__global__ __launch_bounds__(32 * WAVES)
void node_k(const float* __restrict__ mi, const unsigned short* __restrict__ fr,
            const float* __restrict__ hb1, const float* __restrict__ hb2,
            float* __restrict__ h, unsigned short* __restrict__ hbf, int N, int iters)
{
    __shared__ __align__(16) unsigned short wfrag[24 * 512];          // 24 KB (fids 32..55)
    __shared__ __align__(16) unsigned short tile_s[WAVES][16 * HD];

    const int wave = threadIdx.x >> 5;
    const int lane = threadIdx.x & 31;
    const int half = lane >> 4;
    const int m    = lane & 15;

    // stage node-MLP weights: global fids 32..55 -> LDS fids 0..23
    {
        const uint4* s4 = reinterpret_cast<const uint4*>(fr + 32 * 512);
        uint4* d4 = reinterpret_cast<uint4*>(wfrag);
        #pragma unroll
        for (int i = 0; i < (24 * 512 / 8) / (32 * WAVES); ++i)
            d4[threadIdx.x + i * 32 * WAVES] = s4[threadIdx.x + i * 32 * WAVES];
    }

    float c_hb1[4], c_hb2[4];
    #pragma unroll
    for (int nb = 0; nb < 4; ++nb) {
        c_hb1[nb] = hb1[nb * 16 + m];
        c_hb2[nb] = hb2[nb * 16 + m];
    }

    __syncthreads();

    const int stride = gridDim.x * WAVES;
    const int wtile0 = blockIdx.x * WAVES + wave;

    for (int it = 0; it < iters; ++it) {
        const int tile = wtile0 + it * stride;
        const int node = tile * 16 + m;
        const int nc   = (node < N) ? node : 0;

        FragU A[4];
        const uint4* hp = reinterpret_cast<const uint4*>(hbf + (size_t)nc * HD);
        #pragma unroll
        for (int kc = 0; kc < 2; ++kc) {
            A[kc].q[0] = hp[kc * 4 + half];
            A[kc].q[1] = hp[kc * 4 + half + 2];
        }
        const float* mp = mi + (size_t)nc * HD;
        #pragma unroll
        for (int kc = 0; kc < 2; ++kc) {
            #pragma unroll
            for (int i = 0; i < 8; ++i) A[2 + kc].s[i]     = f2bf(mp[kc * 32 + 8 * half + i]);
            #pragma unroll
            for (int i = 0; i < 8; ++i) A[2 + kc].s[8 + i] = f2bf(mp[kc * 32 + 16 + 8 * half + i]);
        }

        #pragma unroll
        for (int nb = 0; nb < 4; ++nb) {
            v8f acc;
            #pragma unroll
            for (int v = 0; v < 8; ++v) acc[v] = c_hb1[nb];
            acc = wmma_bf16(A[0].v, load_bfrag(wfrag,  0 + nb, lane).v, acc);
            acc = wmma_bf16(A[1].v, load_bfrag(wfrag,  4 + nb, lane).v, acc);
            acc = wmma_bf16(A[2].v, load_bfrag(wfrag,  8 + nb, lane).v, acc);
            acc = wmma_bf16(A[3].v, load_bfrag(wfrag, 12 + nb, lane).v, acc);
            #pragma unroll
            for (int v = 0; v < 8; ++v)
                tile_s[wave][(v + 8 * half) * HD + nb * 16 + m] = f2bf(silu_f(acc[v]));
        }
        WB();

        FragU At[2];
        const uint4* ts = reinterpret_cast<const uint4*>(&tile_s[wave][m * HD]);
        #pragma unroll
        for (int kc = 0; kc < 2; ++kc) {
            At[kc].q[0] = ts[kc * 4 + half];
            At[kc].q[1] = ts[kc * 4 + half + 2];
        }
        #pragma unroll
        for (int nb = 0; nb < 4; ++nb) {
            v8f acc;
            #pragma unroll
            for (int v = 0; v < 8; ++v) acc[v] = c_hb2[nb];
            acc = wmma_bf16(At[0].v, load_bfrag(wfrag, 16 + nb, lane).v, acc);
            acc = wmma_bf16(At[1].v, load_bfrag(wfrag, 20 + nb, lane).v, acc);
            #pragma unroll
            for (int v = 0; v < 8; ++v) {
                const int M  = v + 8 * half;
                const int nd = tile * 16 + M;
                if (nd < N) {
                    const size_t idx = (size_t)nd * HD + nb * 16 + m;
                    const float hv = h[idx] + acc[v];
                    h[idx]   = hv;
                    hbf[idx] = f2bf(hv);
                }
            }
        }
        WB();
    }
}

// ---------- finalize: d_out = [h (N*64) | x (N*3)] ----------

__global__ void final_k(const float* __restrict__ h, const float* __restrict__ x,
                        float* __restrict__ out, int N)
{
    const int i = blockIdx.x * blockDim.x + threadIdx.x;
    if (i < N * HD) out[i] = h[i];
    if (i < N * 3)  out[(size_t)N * HD + i] = x[i];
}

// ---------- host ----------

extern "C" void kernel_launch(void* const* d_in, const int* in_sizes, int n_in,
                              void* d_out, int out_size, void* d_ws, size_t ws_size,
                              hipStream_t stream) {
    (void)n_in; (void)out_size; (void)ws_size;

    const int*   an   = (const int*)  d_in[0];
    const float* pos  = (const float*)d_in[1];
    const int*   ei   = (const int*)  d_in[2];
    /* d_in[3] edge_attr: unused by reference */
    const float* emb  = (const float*)d_in[4];
    const float* e_w1 = (const float*)d_in[5];
    const float* e_b1 = (const float*)d_in[6];
    const float* e_w2 = (const float*)d_in[7];
    const float* e_b2 = (const float*)d_in[8];
    const float* h_w1 = (const float*)d_in[9];
    const float* h_b1 = (const float*)d_in[10];
    const float* h_w2 = (const float*)d_in[11];
    const float* h_b2 = (const float*)d_in[12];
    const float* x_w1 = (const float*)d_in[13];
    const float* x_b1 = (const float*)d_in[14];
    const float* x_w2 = (const float*)d_in[15];
    const float* x_b2 = (const float*)d_in[16];

    const int N = in_sizes[1] / 3;
    const int E = in_sizes[2] / 2;

    // workspace carve-up (256B aligned)
    size_t off = 0;
    auto carve = [&](size_t bytes) -> size_t {
        size_t o = off;
        off = (off + bytes + 255) & ~(size_t)255;
        return o;
    };
    char* ws = (char*)d_ws;
    float*          h_buf   = (float*)         (ws + carve((size_t)N * HD * 4));
    unsigned short* hbf_buf = (unsigned short*)(ws + carve((size_t)N * HD * 2));
    float*          mi_buf  = (float*)         (ws + carve((size_t)N * HD * 4));
    float*          xA      = (float*)         (ws + carve((size_t)N * 3 * 4));
    float*          xB      = (float*)         (ws + carve((size_t)N * 3 * 4));
    unsigned short* frags   = (unsigned short*)(ws + carve((size_t)NLAYERS * 56 * 512 * 2));
    float*          ew1last = (float*)         (ws + carve((size_t)NLAYERS * HD * 4));
    float*          xw2buf  = (float*)         (ws + carve((size_t)NLAYERS * HD * 4));

    // pack weights to bf16 B-fragments
    prep_weights_k<<<NLAYERS * 56, 256, 0, stream>>>(e_w1, e_w2, x_w1, h_w1, h_w2, frags);
    prep_small_k<<<1, 128, 0, stream>>>(e_w1, x_w2, ew1last, xw2buf);

    // embedding lookup + x init
    embed_k<<<(N * HD + 255) / 256, 256, 0, stream>>>(an, emb, pos, h_buf, hbf_buf, xA, N);

    const int etiles  = (E + 15) / 16;
    int eblocks = (etiles + WAVES - 1) / WAVES;
    if (eblocks > MAXBLK) eblocks = MAXBLK;
    const int eiters  = (etiles + eblocks * WAVES - 1) / (eblocks * WAVES);

    const int ntiles  = (N + 15) / 16;
    int nblocks = (ntiles + WAVES - 1) / WAVES;
    if (nblocks > MAXBLK) nblocks = MAXBLK;
    const int niters  = (ntiles + nblocks * WAVES - 1) / (nblocks * WAVES);

    for (int l = 0; l < NLAYERS; ++l) {
        float* xcur  = (l == 0) ? xA : xB;
        float* xnext = (l == 0) ? xB : xA;
        const unsigned short* fr_l = frags + (size_t)l * 56 * 512;

        init_layer_k<<<(N * HD + 255) / 256, 256, 0, stream>>>(mi_buf, xcur, xnext, N);

        edge_k<<<eblocks, 32 * WAVES, 0, stream>>>(
            ei, xcur, hbf_buf, fr_l,
            e_b1 + (size_t)l * HD, e_b2 + (size_t)l * HD, x_b1 + (size_t)l * HD,
            ew1last + (size_t)l * HD, xw2buf + (size_t)l * HD, x_b2 + l,
            xnext, mi_buf, E, eiters);

        node_k<<<nblocks, 32 * WAVES, 0, stream>>>(
            mi_buf, fr_l, h_b1 + (size_t)l * HD, h_b2 + (size_t)l * HD,
            h_buf, hbf_buf, N, niters);
    }

    // final x lives in xA after layer 1 (A -> B -> A)
    final_k<<<(N * HD + 255) / 256, 256, 0, stream>>>(h_buf, xA, (float*)d_out, N);
}
